// SparseBottleneck_25262997635675
// MI455X (gfx1250) — matrix operand order repaired and checked by reading, not compiled, not run on hardware
//
#include <hip/hip_runtime.h>
#include <hip/hip_bf16.h>
#include <hip/hip_fp16.h>

typedef __attribute__((ext_vector_type(16))) _Float16 v16h;
typedef __attribute__((ext_vector_type(2)))  _Float16 v2h;
typedef __attribute__((ext_vector_type(8)))  float    v8f;

#define NSITES 500000
#define NC1 128
#define NCH 64
#define NC2 128
#define EPSV 1e-5f
#define NTILES (NSITES / 16)   // 31250 exactly

// fast SiLU: x * rcp(1 + exp(-x))  -> v_exp_f32 + v_rcp_f32 + fma/mul
__device__ __forceinline__ float silu_f(float x) {
    return x * __builtin_amdgcn_rcpf(1.0f + __expf(-x));
}

// ---------------------------------------------------------------------------
// Kernel 1: h = silu(bn(features @ W1))   [N,128] x [128,64] -> [N,64] f16
// ---------------------------------------------------------------------------
__global__ __launch_bounds__(256)
void cv1_kernel(const float* __restrict__ features,
                const float* __restrict__ W1,
                const float* __restrict__ g1, const float* __restrict__ b1,
                const float* __restrict__ m1, const float* __restrict__ v1,
                _Float16* __restrict__ hp, int nblocks)
{
    // W1 pre-packed in WMMA B-operand order: op = kb*4+nb, [op][lane][half]
    __shared__ __align__(32) _Float16 sB[16 * 32 * 16];           // 16 KB
    // Per-wave A tile in WMMA A-operand order: [kb][lane][half]
    __shared__ __align__(32) _Float16 sA[8][4 * 32 * 16];         // 32 KB

    const int tid  = threadIdx.x;
    const int lane = tid & 31;
    const int w    = tid >> 5;

    // pack W1 (f32 -> f16) into operand layout
    for (int i = tid; i < 16 * 32 * 16; i += 256) {
        int half = i & 15, ln = (i >> 4) & 31, op = i >> 9;
        int kb = op >> 2, nb = op & 3;
        int k = kb * 32 + ((ln >> 4) << 3) + (half & 7) + ((half >> 3) << 4);   // B: lanes>=16 hold K+8 (table order)
        int n = nb * 16 + (ln & 15);
        sB[i] = (_Float16)W1[k * NCH + n];
    }
    __syncthreads();

    // zero sentinel row hp[N]
    if (blockIdx.x == 0 && tid < NCH) hp[(size_t)NSITES * NCH + tid] = (_Float16)0.0f;

    // folded BN constants for this lane's 4 output columns
    float scale[4], shift[4];
    for (int nb = 0; nb < 4; ++nb) {
        int c = nb * 16 + (lane & 15);
        float s = g1[c] * rsqrtf(v1[c] + EPSV);
        scale[nb] = s;
        shift[nb] = b1[c] - m1[c] * s;
    }

    // B operands live in registers for the whole kernel (16 x v16h)
    v16h Breg[16];
#pragma unroll
    for (int op = 0; op < 16; ++op)
        Breg[op] = *(const v16h*)&sB[(op * 32 + lane) * 16];

    const int wave_g = blockIdx.x * 8 + w;
    const int wstride = nblocks * 8;

    for (int tile = wave_g; tile < NTILES; tile += wstride) {
        const int row0 = tile * 16;
        // stage 16x128 f32 tile -> f16 in WMMA A layout, 2 K-values at a time
        // (pair k, k+1 lands in adjacent halves of the same lane)
        for (int i = lane; i < 16 * (NC1 / 2); i += 32) {
            int m = i >> 6, k = (i & 63) * 2;
            int kb = k >> 5, kk = k & 31;
            int hi = kk >> 3;
            int ln = m + ((hi & 1) << 4);
            int hf = (kk & 7) + ((hi >> 1) << 3);
            float2 f = *(const float2*)&features[(size_t)(row0 + m) * NC1 + k];
            v2h hv; hv[0] = (_Float16)f.x; hv[1] = (_Float16)f.y;   // v_cvt_pk_f16_f32
            *(v2h*)&sA[w][(kb * 32 + ln) * 16 + hf] = hv;           // ds_store_b32
        }
        asm volatile("s_wait_dscnt 0" ::: "memory");
        __builtin_amdgcn_wave_barrier();

        v16h a[4];
#pragma unroll
        for (int kb = 0; kb < 4; ++kb)
            a[kb] = *(const v16h*)&sA[w][(kb * 32 + lane) * 16];

#pragma unroll
        for (int nb = 0; nb < 4; ++nb) {
            v8f acc = {};
#pragma unroll
            for (int kb = 0; kb < 4; ++kb)
                acc = __builtin_amdgcn_wmma_f32_16x16x32_f16(
                    false, a[kb], false, Breg[kb * 4 + nb], (short)0, acc, false, false);
            int c = nb * 16 + (lane & 15);
            int rbase = row0 + ((lane >> 4) << 3);
#pragma unroll
            for (int r = 0; r < 8; ++r) {
                float x = acc[r] * scale[nb] + shift[nb];
                hp[(size_t)(rbase + r) * NCH + c] = (_Float16)silu_f(x);
            }
        }
        __builtin_amdgcn_wave_barrier();
    }
}

// ---------------------------------------------------------------------------
// Kernel 2: out = silu(bn2( sum_k gather(hp, nbr[k]) @ W2[k] ))  -> h2 f16
// ---------------------------------------------------------------------------
__global__ __launch_bounds__(256)
void cv2_kernel(const _Float16* __restrict__ hp,
                const int* __restrict__ nbr,            // [9][N]
                const float* __restrict__ W2,           // [9][64][64]
                const float* __restrict__ g2, const float* __restrict__ b2,
                const float* __restrict__ m2, const float* __restrict__ v2,
                _Float16* __restrict__ h2, int nblocks)
{
    // 9 taps x 2 kchunks x 4 ntiles = 72 B operands pre-packed: 72 KB
    __shared__ __align__(32) _Float16 sB[72 * 32 * 16];
    // per-wave gathered A tile (16x64) in WMMA layout: 2 KB each
    __shared__ __align__(32) _Float16 sA[8][2 * 32 * 16];

    const int tid  = threadIdx.x;
    const int lane = tid & 31;
    const int w    = tid >> 5;

    for (int i = tid; i < 72 * 32 * 16; i += 256) {
        int half = i & 15, ln = (i >> 4) & 31, op = i >> 9;
        int k9 = op >> 3, kb = (op >> 2) & 1, nb = op & 3;
        int kidx = kb * 32 + ((ln >> 4) << 3) + (half & 7) + ((half >> 3) << 4);
        int n = nb * 16 + (ln & 15);
        sB[i] = (_Float16)W2[((size_t)k9 * NCH + kidx) * NCH + n];
    }
    __syncthreads();

    float scale[4], shift[4];
    for (int nb = 0; nb < 4; ++nb) {
        int c = nb * 16 + (lane & 15);
        float s = g2[c] * rsqrtf(v2[c] + EPSV);
        scale[nb] = s;
        shift[nb] = b2[c] - m2[c] * s;
    }

    const int wave_g = blockIdx.x * 8 + w;
    const int wstride = nblocks * 8;

    for (int tile = wave_g; tile < NTILES; tile += wstride) {
        const int row0 = tile * 16;
        v8f acc[4] = {{}, {}, {}, {}};

        for (int k9 = 0; k9 < 9; ++k9) {
            // rulebook gather: 16 rows x 64 ch f16, paired b32 moves;
            // lane L handles column pair 2L, rows m = 0..15
            for (int i = lane; i < 16 * (NCH / 2); i += 32) {
                int m = i >> 5, c = (i & 31) * 2;
                int src = nbr[k9 * NSITES + row0 + m];   // wave-uniform, coalesces
                v2h val = *(const v2h*)&hp[(size_t)src * NCH + c];
                int kb = c >> 5, kk = c & 31;
                int hi = kk >> 3;
                int ln = m + ((hi & 1) << 4);
                int hf = (kk & 7) + ((hi >> 1) << 3);
                *(v2h*)&sA[w][(kb * 32 + ln) * 16 + hf] = val;
            }
            asm volatile("s_wait_dscnt 0" ::: "memory");
            __builtin_amdgcn_wave_barrier();

            v16h a0 = *(const v16h*)&sA[w][(0 * 32 + lane) * 16];
            v16h a1 = *(const v16h*)&sA[w][(1 * 32 + lane) * 16];

#pragma unroll
            for (int nb = 0; nb < 4; ++nb) {
                v16h b0 = *(const v16h*)&sB[(((k9 * 8) + 0 + nb) * 32 + lane) * 16];
                v16h b1 = *(const v16h*)&sB[(((k9 * 8) + 4 + nb) * 32 + lane) * 16];
                acc[nb] = __builtin_amdgcn_wmma_f32_16x16x32_f16(
                    false, a0, false, b0, (short)0, acc[nb], false, false);
                acc[nb] = __builtin_amdgcn_wmma_f32_16x16x32_f16(
                    false, a1, false, b1, (short)0, acc[nb], false, false);
            }
            __builtin_amdgcn_wave_barrier();
        }

#pragma unroll
        for (int nb = 0; nb < 4; ++nb) {
            int c = nb * 16 + (lane & 15);
            int rbase = row0 + ((lane >> 4) << 3);
#pragma unroll
            for (int r = 0; r < 8; ++r) {
                float x = acc[nb][r] * scale[nb] + shift[nb];
                h2[(size_t)(rbase + r) * NCH + c] = (_Float16)silu_f(x);
            }
        }
    }
}

// ---------------------------------------------------------------------------
// Kernel 3: out = silu(bn3(h2 @ W3) + features)   [N,64]x[64,128] -> [N,128]
// ---------------------------------------------------------------------------
__global__ __launch_bounds__(256)
void cv3_kernel(const _Float16* __restrict__ h2,
                const float* __restrict__ features,
                const float* __restrict__ W3,           // [64,128]
                const float* __restrict__ g3, const float* __restrict__ b3,
                const float* __restrict__ m3, const float* __restrict__ v3,
                float* __restrict__ out, int nblocks)
{
    __shared__ __align__(32) _Float16 sB[16 * 32 * 16];   // 2 kb x 8 nb = 16 ops
    __shared__ __align__(32) _Float16 sA[8][2 * 32 * 16];

    const int tid  = threadIdx.x;
    const int lane = tid & 31;
    const int w    = tid >> 5;

    for (int i = tid; i < 16 * 32 * 16; i += 256) {
        int half = i & 15, ln = (i >> 4) & 31, op = i >> 9;
        int kb = op >> 3, nb = op & 7;
        int k = kb * 32 + ((ln >> 4) << 3) + (half & 7) + ((half >> 3) << 4);
        int n = nb * 16 + (ln & 15);
        sB[i] = (_Float16)W3[k * NC2 + n];
    }
    __syncthreads();

    float scale[8], shift[8];
    for (int nb = 0; nb < 8; ++nb) {
        int c = nb * 16 + (lane & 15);
        float s = g3[c] * rsqrtf(v3[c] + EPSV);
        scale[nb] = s;
        shift[nb] = b3[c] - m3[c] * s;
    }

    v16h Breg[16];
#pragma unroll
    for (int op = 0; op < 16; ++op)
        Breg[op] = *(const v16h*)&sB[(op * 32 + lane) * 16];

    const int wave_g = blockIdx.x * 8 + w;
    const int wstride = nblocks * 8;

    for (int tile = wave_g; tile < NTILES; tile += wstride) {
        const int row0 = tile * 16;
        // stage 16x64 f16 tile with paired b32 copies
        for (int i = lane; i < 16 * (NCH / 2); i += 32) {
            int m = i >> 5, c = (i & 31) * 2;
            int kb = c >> 5, kk = c & 31;
            int hi = kk >> 3;
            int ln = m + ((hi & 1) << 4);
            int hf = (kk & 7) + ((hi >> 1) << 3);
            *(v2h*)&sA[w][(kb * 32 + ln) * 16 + hf] =
                *(const v2h*)&h2[(size_t)(row0 + m) * NCH + c];
        }
        asm volatile("s_wait_dscnt 0" ::: "memory");
        __builtin_amdgcn_wave_barrier();

        v16h a0 = *(const v16h*)&sA[w][(0 * 32 + lane) * 16];
        v16h a1 = *(const v16h*)&sA[w][(1 * 32 + lane) * 16];

#pragma unroll
        for (int nb = 0; nb < 8; ++nb) {
            v8f acc = {};
            acc = __builtin_amdgcn_wmma_f32_16x16x32_f16(
                false, a0, false, Breg[0 * 8 + nb], (short)0, acc, false, false);
            acc = __builtin_amdgcn_wmma_f32_16x16x32_f16(
                false, a1, false, Breg[1 * 8 + nb], (short)0, acc, false, false);
            int c = nb * 16 + (lane & 15);
            int rbase = row0 + ((lane >> 4) << 3);
#pragma unroll
            for (int r = 0; r < 8; ++r) {
                int row = rbase + r;
                float x = acc[r] * scale[nb] + shift[nb] + features[(size_t)row * NC2 + c];
                out[(size_t)row * NC2 + c] = silu_f(x);
            }
        }
        __builtin_amdgcn_wave_barrier();
    }
}

// ---------------------------------------------------------------------------
extern "C" void kernel_launch(void* const* d_in, const int* in_sizes, int n_in,
                              void* d_out, int out_size, void* d_ws, size_t ws_size,
                              hipStream_t stream) {
    (void)in_sizes; (void)n_in; (void)out_size; (void)ws_size;
    const float* features = (const float*)d_in[0];
    const int*   nbr      = (const int*)d_in[1];
    const float* W1 = (const float*)d_in[2];
    const float* W2 = (const float*)d_in[3];
    const float* W3 = (const float*)d_in[4];
    const float* g1 = (const float*)d_in[5];
    const float* b1 = (const float*)d_in[6];
    const float* m1 = (const float*)d_in[7];
    const float* v1 = (const float*)d_in[8];
    const float* g2 = (const float*)d_in[9];
    const float* b2 = (const float*)d_in[10];
    const float* m2 = (const float*)d_in[11];
    const float* v2 = (const float*)d_in[12];
    const float* g3 = (const float*)d_in[13];
    const float* b3 = (const float*)d_in[14];
    const float* m3 = (const float*)d_in[15];
    const float* v3 = (const float*)d_in[16];

    // workspace: hp [(N+1) x 64] f16, then h2 [N x 64] f16  (~122 MB total)
    _Float16* hp = (_Float16*)d_ws;
    _Float16* h2 = hp + (size_t)(NSITES + 1) * NCH;

    const int blocks = 1024;
    cv1_kernel<<<blocks, 256, 0, stream>>>(features, W1, g1, b1, m1, v1, hp, blocks);
    cv2_kernel<<<blocks, 256, 0, stream>>>(hp, nbr, W2, g2, b2, m2, v2, h2, blocks);
    cv3_kernel<<<blocks, 256, 0, stream>>>(h2, features, W3, g3, b3, m3, v3,
                                           (float*)d_out, blocks);
}
